// AligningModel_77661598646786
// MI455X (gfx1250) — compile-verified, matrix operand order checked by reading, not run on hardware
//
#include <hip/hip_runtime.h>

// ---------------- CDNA5 WMMA types ----------------
typedef __attribute__((ext_vector_type(16))) __bf16 v16bf;
typedef __attribute__((ext_vector_type(8)))  float  v8f;

#define E_     256
#define MEL_   80
#define VOCAB_ 100
#define B_     16
#define M_     2000
#define P_     400

static __device__ __forceinline__ __bf16 f2bf(float f) {
  unsigned u = __builtin_bit_cast(unsigned, f);
  unsigned r = (u + 0x7FFFu + ((u >> 16) & 1u)) >> 16;
  unsigned short s = (unsigned short)r;
  return __builtin_bit_cast(__bf16, s);
}
static __device__ __forceinline__ float bf2f(__bf16 b) {
  unsigned short s = __builtin_bit_cast(unsigned short, b);
  unsigned u = ((unsigned)s) << 16;
  return __builtin_bit_cast(float, u);
}
static __device__ __forceinline__ float sigm(float x) { return 1.0f / (1.0f + __expf(-x)); }

union FragU { v16bf v; uint4 q[2]; };

// Load one 16-lane-striped WMMA operand fragment (A rows, or B given as
// [N][K] row-major) from a row-major bf16 matrix.  rowptr already includes
// (row0 + (lane&15)) * stride.  K chunk pattern per CDNA5 ISA:
//   j in 0..7  -> K = j + 8*half ;  j in 8..15 -> K = j + 8 + 8*half
static __device__ __forceinline__ v16bf load_frag(const __bf16* rowptr, int k0, int hl) {
  FragU f;
  f.q[0] = *(const uint4*)(rowptr + k0 + 8 * hl);
  f.q[1] = *(const uint4*)(rowptr + k0 + 16 + 8 * hl);
  return f.v;
}

// =====================================================================
// Generic bf16 WMMA GEMM:  C[M,N] (+=) A[M,K] * Bt[N,K]^T  (+ bias[n])
// Block = 256 threads = 8 waves (4x2), tile 128(M) x 64(N), K step 32.
// =====================================================================
__global__ __launch_bounds__(256) void k_gemm(
    const __bf16* __restrict__ A,  long sA, int lda,
    const __bf16* __restrict__ Bt, long sB, int ldb,
    float* __restrict__ C,         long sC, int ldc,
    const float* __restrict__ bias,
    int M, int N, int K, int accum) {
  __shared__ __align__(16) __bf16 As[128 * 40];
  __shared__ __align__(16) __bf16 Bs[64 * 40];
  A  += (long)blockIdx.z * sA;
  Bt += (long)blockIdx.z * sB;
  C  += (long)blockIdx.z * sC;
  const int m0 = blockIdx.y * 128, n0 = blockIdx.x * 64;
  const int tid = threadIdx.x, lane = tid & 31, wv = tid >> 5;
  const int wm = wv >> 1, wn = wv & 1;
  const int hl = lane >> 4, ln = lane & 15;

  v8f acc00 = {}, acc01 = {}, acc10 = {}, acc11 = {};
  const int kt = (K + 31) >> 5;
  for (int kb = 0; kb < kt; ++kb) {
    const int k0 = kb << 5;
    __syncthreads();
    {  // stage A tile: 128 rows x 32 cols
      const int ar = tid >> 1, kh = (tid & 1) << 4;
      const int gr = m0 + ar;
      const __bf16* ap = A + (size_t)gr * lda + k0 + kh;
      __bf16* d = As + ar * 40 + kh;
      if (gr < M && (k0 + kh + 16) <= K) {
        *(uint4*)d       = *(const uint4*)ap;
        *(uint4*)(d + 8) = *(const uint4*)(ap + 8);
      } else {
        for (int i = 0; i < 16; ++i) {
          const int kk = k0 + kh + i;
          d[i] = (gr < M && kk < K) ? ap[i] : f2bf(0.f);
        }
      }
      if (kb + 1 < kt && gr < M) __builtin_prefetch(ap + 32, 0, 0);
    }
    {  // stage Bt tile: 64 rows(n) x 32 cols(k)
      const int br = tid >> 2, kq = (tid & 3) << 3;
      const int gn = n0 + br;
      const __bf16* bp = Bt + (size_t)gn * ldb + k0 + kq;
      __bf16* d = Bs + br * 40 + kq;
      if (gn < N && (k0 + kq + 8) <= K) {
        *(uint4*)d = *(const uint4*)bp;
      } else {
        for (int i = 0; i < 8; ++i) {
          const int kk = k0 + kq + i;
          d[i] = (gn < N && kk < K) ? bp[i] : f2bf(0.f);
        }
      }
    }
    __syncthreads();
    const __bf16* ar0 = As + (wm * 32 + ln) * 40;
    const __bf16* ar1 = ar0 + 16 * 40;
    const __bf16* br0 = Bs + (wn * 32 + ln) * 40;
    const __bf16* br1 = br0 + 16 * 40;
    const v16bf a0 = load_frag(ar0, 0, hl);
    const v16bf a1 = load_frag(ar1, 0, hl);
    const v16bf b0 = load_frag(br0, 0, hl);
    const v16bf b1 = load_frag(br1, 0, hl);
    acc00 = __builtin_amdgcn_wmma_f32_16x16x32_bf16(false, a0, false, b0, (short)0, acc00, false, false);
    acc01 = __builtin_amdgcn_wmma_f32_16x16x32_bf16(false, a0, false, b1, (short)0, acc01, false, false);
    acc10 = __builtin_amdgcn_wmma_f32_16x16x32_bf16(false, a1, false, b0, (short)0, acc10, false, false);
    acc11 = __builtin_amdgcn_wmma_f32_16x16x32_bf16(false, a1, false, b1, (short)0, acc11, false, false);
  }
  // epilogue: lane covers n = base + (lane&15); rows m = base + r + 8*half
  const int mb = m0 + wm * 32, nb = n0 + wn * 32;
  auto st = [&](v8f& acc, int mt, int nt) {
    const int n = nb + nt + ln;
    if (n >= N) return;
    const float bv = bias ? bias[n] : 0.f;
    for (int r = 0; r < 8; ++r) {
      const int m = mb + mt + r + 8 * hl;
      if (m < M) {
        const size_t idx = (size_t)m * ldc + n;
        float v = acc[r] + bv;
        if (accum) v += C[idx];
        C[idx] = v;
      }
    }
  };
  st(acc00, 0, 0); st(acc01, 0, 16); st(acc10, 16, 0); st(acc11, 16, 16);
}

// =====================================================================
// Persistent LSTM recurrence: one workgroup, 16 waves, B=16 batch rows
// fill the WMMA M dim.  h kept in LDS (bf16), c in registers.  Per step
// gates = pre_gates[b,t] + h @ Whh^T via v_wmma (K=256, N=1024).
// Whh passed in torch [4H, H] = [n][k] layout (fragment-native).
// =====================================================================
__global__ __launch_bounds__(512) void k_lstm_rec(
    const float* __restrict__ preg,   // [B*T, 1024], row = b*T + t (bias included)
    const __bf16* __restrict__ Whh,   // [1024, 256] bf16
    __bf16* __restrict__ out,         // [B*T, ldo], write h at col offo+j
    int T, int ldo, int offo, int rev) {
  __shared__ __align__(16) __bf16 hb[16 * 256];
  const int tid = threadIdx.x, lane = tid & 31, wv = tid >> 5;  // 16 waves
  const int hl = lane >> 4, ln = lane & 15;
  for (int i = tid; i < 16 * 256; i += 512) hb[i] = f2bf(0.f);
  float c[8];
  for (int r = 0; r < 8; ++r) c[r] = 0.f;
  const int j = wv * 16 + ln;              // hidden unit column for this lane
  const __bf16* arow = hb + ln * 256;      // A row (batch = lane&15)
  __syncthreads();

  for (int s = 0; s < T; ++s) {
    const int t = rev ? (T - 1 - s) : s;
    v16bf a[8];
#pragma unroll
    for (int kk = 0; kk < 8; ++kk) a[kk] = load_frag(arow, kk * 32, hl);
    v8f acc[4] = {};
#pragma unroll
    for (int g = 0; g < 4; ++g) {
      const __bf16* brow = Whh + (size_t)(g * 256 + wv * 16 + ln) * 256;
#pragma unroll
      for (int kk = 0; kk < 8; ++kk) {
        const v16bf b = load_frag(brow, kk * 32, hl);
        acc[g] = __builtin_amdgcn_wmma_f32_16x16x32_bf16(false, a[kk], false, b, (short)0, acc[g], false, false);
      }
    }
    __syncthreads();  // everyone finished reading hb
#pragma unroll
    for (int r = 0; r < 8; ++r) {
      const int bb = r + 8 * hl;           // batch index
      const size_t row = ((size_t)bb * T + t) * 1024;
      const float gi = preg[row + j]       + acc[0][r];
      const float gf = preg[row + 256 + j] + acc[1][r];
      const float gg = preg[row + 512 + j] + acc[2][r];
      const float go = preg[row + 768 + j] + acc[3][r];
      const float cc = sigm(gf) * c[r] + sigm(gi) * tanhf(gg);
      const float h  = sigm(go) * tanhf(cc);
      c[r] = cc;
      const __bf16 hv = f2bf(h);
      hb[bb * 256 + j] = hv;
      out[((size_t)bb * T + t) * ldo + offo + j] = hv;
    }
    __syncthreads();  // hb updated before next step's A loads
  }
}

// ---------------- elementwise / data-movement kernels ----------------
__global__ void k_cvt(const float* __restrict__ in, __bf16* __restrict__ out, long n) {
  long i = (long)blockIdx.x * 256 + threadIdx.x;
  if (i < n) out[i] = f2bf(in[i]);
}

__global__ void k_shift_mask(const float* __restrict__ enc, const int* __restrict__ mlen,
                             __bf16* __restrict__ ash, int off) {
  long idx = (long)blockIdx.x * 256 + threadIdx.x;
  if (idx >= (long)B_ * M_ * 256) return;
  const int e = idx & 255;
  const long row = idx >> 8;
  const int t = (int)(row % M_), b = (int)(row / M_);
  const int ts = t + off;
  float v = 0.f;
  if (ts >= 0 && ts < M_ && ts < mlen[b]) v = enc[((size_t)b * M_ + ts) * 256 + e];
  ash[idx] = f2bf(v);
}

__global__ void k_prep_conv(const float* __restrict__ W, __bf16* __restrict__ out, int ks, int j) {
  long idx = (long)blockIdx.x * 256 + threadIdx.x;
  if (idx < 512L * 256) out[idx] = f2bf(W[idx * ks + j]);  // W: [512][256][ks]
}

__global__ void k_glu(float* __restrict__ enc, const float* __restrict__ y) {
  long idx = (long)blockIdx.x * 256 + threadIdx.x;
  if (idx >= (long)B_ * M_ * 256) return;
  const int e = idx & 255;
  const long row = idx >> 8;
  const float a = y[row * 512 + e], g = y[row * 512 + 256 + e];
  enc[idx] += a * sigm(g);
}

__global__ void k_embed(const int* __restrict__ tok, const float* __restrict__ emb,
                        __bf16* __restrict__ out) {
  long idx = (long)blockIdx.x * 256 + threadIdx.x;
  if (idx >= (long)B_ * P_ * 256) return;
  const int e = idx & 255;
  const long row = idx >> 8;
  out[idx] = f2bf(emb[(size_t)tok[row] * 256 + e]);
}

__global__ __launch_bounds__(128) void k_norm(const __bf16* __restrict__ x,
                                              float* __restrict__ nrm, int width) {
  const int row = blockIdx.x;
  const __bf16* p = x + (size_t)row * width;
  float s = 0.f;
  for (int i = threadIdx.x; i < width; i += 128) { const float v = bf2f(p[i]); s += v * v; }
  __shared__ float red[128];
  red[threadIdx.x] = s; __syncthreads();
  for (int o = 64; o > 0; o >>= 1) { if (threadIdx.x < o) red[threadIdx.x] += red[threadIdx.x + o]; __syncthreads(); }
  if (threadIdx.x == 0) nrm[row] = red[0];
}

// in-place: align region holds mel_enc . ph_enc^T dot products; turn into softmax(-dist)
__global__ __launch_bounds__(128) void k_softmax(float* __restrict__ alg,
                                                 const float* __restrict__ mn,
                                                 const float* __restrict__ pn,
                                                 const int* __restrict__ mlen,
                                                 const int* __restrict__ plen) {
  const int row = blockIdx.x;                 // = b*M + m
  const int b = row / M_, m = row % M_;
  float* p = alg + (size_t)row * P_;
  const int pl = plen[b], ml = mlen[b];
  const float mnv = mn[row];
  float v[4];
  float mx = -1e30f;
  for (int i = 0; i < 4; ++i) {
    const int pp = threadIdx.x + i * 128;
    float d = -1e9f;
    if (pp < P_ && pp < pl) {
      const float d2 = fmaxf(mnv + pn[b * P_ + pp] - 2.f * p[pp], 1e-12f);
      d = -__builtin_sqrtf(d2);
    }
    v[i] = d;
    if (pp < P_) mx = fmaxf(mx, d);
  }
  __shared__ float red[128];
  red[threadIdx.x] = mx; __syncthreads();
  for (int o = 64; o > 0; o >>= 1) { if (threadIdx.x < o) red[threadIdx.x] = fmaxf(red[threadIdx.x], red[threadIdx.x + o]); __syncthreads(); }
  mx = red[0]; __syncthreads();
  float s = 0.f;
  for (int i = 0; i < 4; ++i) {
    const int pp = threadIdx.x + i * 128;
    if (pp < P_) { v[i] = __expf(v[i] - mx); s += v[i]; }
  }
  red[threadIdx.x] = s; __syncthreads();
  for (int o = 64; o > 0; o >>= 1) { if (threadIdx.x < o) red[threadIdx.x] += red[threadIdx.x + o]; __syncthreads(); }
  s = red[0];
  const float inv = (m < ml) ? (1.f / s) : 0.f;   // zero padded mel rows
  for (int i = 0; i < 4; ++i) {
    const int pp = threadIdx.x + i * 128;
    if (pp < P_) p[pp] = v[i] * inv;
  }
}

__global__ void k_tr_f2b(const float* __restrict__ in, __bf16* __restrict__ out,
                         int R, int C, long total) {  // out[b][c][r] = in[b][r][c]
  long idx = (long)blockIdx.x * 256 + threadIdx.x;
  if (idx >= total) return;
  const long per = (long)R * C;
  const int b = (int)(idx / per);
  const long rm = idx % per;
  const int r = (int)(rm / C), c = (int)(rm % C);
  out[(size_t)b * per + (size_t)c * R + r] = f2bf(in[idx]);
}

__global__ void k_tr_b2b(const __bf16* __restrict__ in, __bf16* __restrict__ out,
                         int R, int C, long total) {
  long idx = (long)blockIdx.x * 256 + threadIdx.x;
  if (idx >= total) return;
  const long per = (long)R * C;
  const int b = (int)(idx / per);
  const long rm = idx % per;
  const int r = (int)(rm / C), c = (int)(rm % C);
  out[(size_t)b * per + (size_t)c * R + r] = in[idx];
}

// ---------------- host side ----------------
static inline unsigned bks(long n) { return (unsigned)((n + 255) / 256); }

static inline void gemm(hipStream_t s,
                        const __bf16* A, long sA, int lda,
                        const __bf16* Bt, long sB, int ldb,
                        float* C, long sC, int ldc, const float* bias,
                        int M, int N, int K, int accum, int batches) {
  dim3 g((N + 63) / 64, (M + 127) / 128, batches);
  k_gemm<<<g, 256, 0, s>>>(A, sA, lda, Bt, sB, ldb, C, sC, ldc, bias, M, N, K, accum);
}

extern "C" void kernel_launch(void* const* d_in, const int* in_sizes, int n_in,
                              void* d_out, int out_size, void* d_ws, size_t ws_size,
                              hipStream_t stream) {
  (void)in_sizes; (void)n_in; (void)out_size; (void)ws_size;
  const float* mels     = (const float*)d_in[0];
  const int*   pwb      = (const int*)d_in[1];
  const int*   mel_lens = (const int*)d_in[3];
  const int*   ph_lens  = (const int*)d_in[4];
  const float* embed    = (const float*)d_in[5];
  const float* phWih[2] = {(const float*)d_in[6],  (const float*)d_in[9]};
  const float* phWhh[2] = {(const float*)d_in[7],  (const float*)d_in[10]};
  const float* phB[2]   = {(const float*)d_in[8],  (const float*)d_in[11]};
  const float* mWih[2]  = {(const float*)d_in[12], (const float*)d_in[15]};
  const float* mWhh[2]  = {(const float*)d_in[13], (const float*)d_in[16]};
  const float* mB[2]    = {(const float*)d_in[14], (const float*)d_in[17]};
  const float* proj_W   = (const float*)d_in[18];
  const float* proj_b   = (const float*)d_in[19];
  const float* convW[3] = {(const float*)d_in[20], (const float*)d_in[22], (const float*)d_in[24]};
  const float* convB[3] = {(const float*)d_in[21], (const float*)d_in[23], (const float*)d_in[25]};
  const float* dec_Wih  = (const float*)d_in[26];
  const float* dec_Whh  = (const float*)d_in[27];
  const float* dec_b    = (const float*)d_in[28];
  const float* out_W    = (const float*)d_in[29];
  const float* out_b    = (const float*)d_in[30];

  float* logits = (float*)d_out;
  float* align  = logits + (size_t)B_ * P_ * VOCAB_;

  char* wp = (char*)d_ws;
  auto carve = [&](size_t bytes) -> void* {
    void* p = (void*)wp;
    wp += (bytes + 255) & ~(size_t)255;
    return p;
  };
  const size_t BM = (size_t)B_ * M_, BP = (size_t)B_ * P_;
  float*  enc     = (float*)carve(BM * 256 * 4);
  float*  ybuf    = (float*)carve(BM * 512 * 4);
  float*  gates   = (float*)carve(BM * 1024 * 4);
  float*  ctx     = (float*)carve(BP * 512 * 4);
  float*  mnorm   = (float*)carve(BM * 4);
  float*  pnorm   = (float*)carve(BP * 4);
  __bf16* melsb   = (__bf16*)carve(BM * MEL_ * 2);
  __bf16* ash     = (__bf16*)carve(BM * 256 * 2);
  __bf16* encb    = (__bf16*)carve(BM * 256 * 2);
  __bf16* melenc  = (__bf16*)carve(BM * 512 * 2);
  __bf16* melencT = (__bf16*)carve(BM * 512 * 2);
  __bf16* phemb   = (__bf16*)carve(BP * 256 * 2);
  __bf16* phenc   = (__bf16*)carve(BP * 512 * 2);
  __bf16* alignT  = (__bf16*)carve((size_t)B_ * P_ * M_ * 2);
  __bf16* ctxb    = (__bf16*)carve(BP * 512 * 2);
  __bf16* hid     = (__bf16*)carve(BP * 256 * 2);
  __bf16* wt      = (__bf16*)carve((size_t)1024 * 512 * 2);

  // --- mel projection: enc = mels @ proj_W^T + proj_b ---
  k_cvt<<<bks(BM * MEL_), 256, 0, stream>>>(mels, melsb, (long)(BM * MEL_));
  k_cvt<<<bks(256 * MEL_), 256, 0, stream>>>(proj_W, wt, 256L * MEL_);
  gemm(stream, melsb, 0, MEL_, wt, 0, MEL_, enc, 0, 256, proj_b, (int)BM, 256, MEL_, 0, 1);

  // --- residual GLU conv blocks as shifted-tap GEMM accumulation ---
  const int ksz[3] = {3, 5, 7};
  for (int blk = 0; blk < 3; ++blk) {
    const int ks = ksz[blk];
    for (int j = 0; j < ks; ++j) {
      const int off = j - ks / 2;
      k_shift_mask<<<bks(BM * 256), 256, 0, stream>>>(enc, mel_lens, ash, off);
      k_prep_conv<<<bks(512L * 256), 256, 0, stream>>>(convW[blk], wt, ks, j);
      gemm(stream, ash, 0, 256, wt, 0, 256, ybuf, 0, 512,
           (j == 0) ? convB[blk] : nullptr, (int)BM, 512, 256, (j != 0), 1);
    }
    k_glu<<<bks(BM * 256), 256, 0, stream>>>(enc, ybuf);
  }
  k_cvt<<<bks(BM * 256), 256, 0, stream>>>(enc, encb, (long)(BM * 256));

  // --- mel BiLSTM ---
  for (int d = 0; d < 2; ++d) {
    k_cvt<<<bks(1024L * 256), 256, 0, stream>>>(mWih[d], wt, 1024L * 256);
    gemm(stream, encb, 0, 256, wt, 0, 256, gates, 0, 1024, mB[d], (int)BM, 1024, 256, 0, 1);
    k_cvt<<<bks(1024L * 256), 256, 0, stream>>>(mWhh[d], wt, 1024L * 256);
    k_lstm_rec<<<1, 512, 0, stream>>>(gates, wt, melenc, M_, 512, d ? 256 : 0, d);
  }

  // --- phoneme BiLSTM ---
  k_embed<<<bks(BP * 256), 256, 0, stream>>>(pwb, embed, phemb);
  for (int d = 0; d < 2; ++d) {
    k_cvt<<<bks(1024L * 256), 256, 0, stream>>>(phWih[d], wt, 1024L * 256);
    gemm(stream, phemb, 0, 256, wt, 0, 256, gates, 0, 1024, phB[d], (int)BP, 1024, 256, 0, 1);
    k_cvt<<<bks(1024L * 256), 256, 0, stream>>>(phWhh[d], wt, 1024L * 256);
    k_lstm_rec<<<1, 512, 0, stream>>>(gates, wt, phenc, P_, 512, d ? 256 : 0, d);
  }

  // --- norms ---
  k_norm<<<(unsigned)BM, 128, 0, stream>>>(melenc, mnorm, 512);
  k_norm<<<(unsigned)BP, 128, 0, stream>>>(phenc, pnorm, 512);

  // --- distance dot products (batched), straight into align region of d_out ---
  gemm(stream, melenc, (long)M_ * 512, 512, phenc, (long)P_ * 512, 512,
       align, (long)M_ * P_, P_, nullptr, M_, P_, 512, 0, B_);
  k_softmax<<<(unsigned)BM, 128, 0, stream>>>(align, mnorm, pnorm, mel_lens, ph_lens);

  // --- context = align^T @ mel_enc (batched) ---
  k_tr_f2b<<<bks((long)B_ * M_ * P_), 256, 0, stream>>>(align, alignT, M_, P_, (long)B_ * M_ * P_);
  k_tr_b2b<<<bks((long)BM * 512), 256, 0, stream>>>(melenc, melencT, M_, 512, (long)BM * 512);
  gemm(stream, alignT, (long)P_ * M_, M_, melencT, (long)512 * M_, M_,
       ctx, (long)P_ * 512, 512, nullptr, P_, 512, M_, 0, B_);
  k_cvt<<<bks(BP * 512), 256, 0, stream>>>(ctx, ctxb, (long)(BP * 512));

  // --- decoder LSTM + output projection ---
  k_cvt<<<bks(1024L * 512), 256, 0, stream>>>(dec_Wih, wt, 1024L * 512);
  gemm(stream, ctxb, 0, 512, wt, 0, 512, gates, 0, 1024, dec_b, (int)BP, 1024, 512, 0, 1);
  k_cvt<<<bks(1024L * 256), 256, 0, stream>>>(dec_Whh, wt, 1024L * 256);
  k_lstm_rec<<<1, 512, 0, stream>>>(gates, wt, hid, P_, 256, 0, 0);

  k_cvt<<<bks((long)VOCAB_ * 256), 256, 0, stream>>>(out_W, wt, (long)VOCAB_ * 256);
  gemm(stream, hid, 0, 256, wt, 0, 256, logits, 0, VOCAB_, out_b, (int)BP, VOCAB_, 256, 0, 1);
}
